// ChromophoreSolventTransformerGNN_10900626997628
// MI455X (gfx1250) — matrix operand order repaired and checked by reading, not compiled, not run on hardware
//
#include <hip/hip_runtime.h>
#include <hip/hip_bf16.h>
#include <math.h>

// ---------------------------------------------------------------------------
// Problem constants (from reference)
// ---------------------------------------------------------------------------
#define NN 32768      // nodes
#define EE 131072     // edges
#define BB 1024       // graphs
#define HH 4          // attention heads
#define BN_EPS 1e-5f
#define MT 4          // m-tiles (16 rows each) per wave in the WMMA GEMM

// ---------------------------------------------------------------------------
// WMMA vector types (gfx1250, wave32)
// ---------------------------------------------------------------------------
typedef __attribute__((ext_vector_type(16))) __bf16         v16bf;
typedef __attribute__((ext_vector_type(8)))  float          v8f;
typedef __attribute__((ext_vector_type(8)))  unsigned short v8us;

union BfFrag { v16bf v; v8us h[2]; };

__device__ __forceinline__ unsigned short f2bf(float f) {
  unsigned u = __float_as_uint(f);
  u += 0x7FFFu + ((u >> 16) & 1u);          // round-to-nearest-even
  return (unsigned short)(u >> 16);
}
__device__ __forceinline__ float bf2f(unsigned short h) {
  return __uint_as_float(((unsigned)h) << 16);
}
// order-preserving float<->uint for atomicMax-based segment max
__device__ __forceinline__ unsigned f2ord(float f) {
  unsigned u = __float_as_uint(f);
  return (u & 0x80000000u) ? ~u : (u | 0x80000000u);
}
__device__ __forceinline__ float ord2f(unsigned u) {
  return (u & 0x80000000u) ? __uint_as_float(u ^ 0x80000000u)
                           : __uint_as_float(~u);
}

// ---------------------------------------------------------------------------
// Pack/cast kernels
// ---------------------------------------------------------------------------
// Wt[n*Kpad + k] = bf16(W[k*N + n]), zero-padded K
__global__ __launch_bounds__(256)
void pack_wt_bf16(const float* __restrict__ W, unsigned short* __restrict__ Wt,
                  int K, int Nn, int Kpad) {
  int idx = blockIdx.x * 256 + threadIdx.x;
  if (idx >= Nn * Kpad) return;
  int n = idx / Kpad, k = idx - n * Kpad;
  Wt[idx] = (k < K) ? f2bf(W[(size_t)k * Nn + n]) : (unsigned short)0;
}

// Xb[m*Kpad + k] = bf16(X[m*K + k]), zero-padded K
__global__ __launch_bounds__(256)
void cast_bf16_pad(const float* __restrict__ X, unsigned short* __restrict__ Xb,
                   int M, int K, int Kpad) {
  int idx = blockIdx.x * 256 + threadIdx.x;
  if (idx >= M * Kpad) return;
  int m = idx / Kpad, k = idx - m * Kpad;
  Xb[idx] = (k < K) ? f2bf(X[(size_t)m * K + k]) : (unsigned short)0;
}

// ---------------------------------------------------------------------------
// WMMA GEMM: C[M,N] = A[M,K](bf16,row-major) x Bt[N,K](bf16,row-major) + bias
// Register-blocked: each wave owns MT=4 vertically-stacked 16x16 C tiles
// (64 rows x 16 cols) and reuses one B fragment across 4 WMMAs per K-step.
// 8 waves per block span 8 adjacent n-tiles (128 cols).
// Requires: K % 32 == 0, M % 64 == 0, N % 16 == 0.
// ---------------------------------------------------------------------------
__global__ __launch_bounds__(256)
void wmma_gemm_bf16(const unsigned short* __restrict__ A,
                    const unsigned short* __restrict__ Bt,
                    const float* __restrict__ bias,
                    float* __restrict__ Cf,            // f32 out (or null)
                    unsigned short* __restrict__ Cb,   // bf16 out (or null)
                    int M, int Nn, int K, int ldc, int relu) {
  const int wave = threadIdx.x >> 5;
  const int lane = threadIdx.x & 31;
  const int half = lane >> 4;
  const int l15  = lane & 15;
  const int n0 = (blockIdx.x * 8 + wave) * 16;
  const int m0 = blockIdx.y * (16 * MT);
  if (n0 >= Nn) return;

  // A fragment (ISA 16-bit A layout): lane<16 holds row m+l15,
  // K = {k0..k0+7, k0+16..k0+23}; lane>=16 shifted +8 in K.
  const unsigned short* Arow = A  + (size_t)(m0 + l15) * K + half * 8;
  // B fragment: lane holds column n0+l15, same K split as A (offset half*8).
  const unsigned short* Brow = Bt + (size_t)(n0 + l15) * K + half * 8;
  const size_t as = (size_t)16 * K;   // row stride between m-tiles

  const v8f vzero = {0.f, 0.f, 0.f, 0.f, 0.f, 0.f, 0.f, 0.f};
  v8f acc[MT];
#pragma unroll
  for (int t = 0; t < MT; ++t) acc[t] = vzero;

  for (int k0 = 0; k0 < K; k0 += 32) {
    // pull next weight cachelines toward the caches (global_prefetch_b8)
    __builtin_prefetch(Brow + k0 + 512, 0, 1);
    BfFrag b;
    b.h[0] = *(const v8us*)(Brow + k0);        // 16B
    b.h[1] = *(const v8us*)(Brow + k0 + 16);   // 16B
    BfFrag a[MT];
#pragma unroll
    for (int t = 0; t < MT; ++t) {
      a[t].h[0] = *(const v8us*)(Arow + t * as + k0);        // 16B
      a[t].h[1] = *(const v8us*)(Arow + t * as + k0 + 16);   // 16B
    }
#pragma unroll
    for (int t = 0; t < MT; ++t)
      acc[t] = __builtin_amdgcn_wmma_f32_16x16x32_bf16(
          /*neg_a=*/false, a[t].v, /*neg_b=*/false, b.v,
          /*c_mod=*/(short)0, acc[t], /*reuse_a=*/false, /*reuse_b=*/false);
  }

  const int n = n0 + l15;
  const float bv = bias ? bias[n] : 0.0f;
#pragma unroll
  for (int t = 0; t < MT; ++t) {
#pragma unroll
    for (int r = 0; r < 8; ++r) {
      float val = acc[t][r] + bv;       // C layout: VGPR r -> row r + half*8
      if (relu) val = fmaxf(val, 0.0f);
      size_t idx = (size_t)(m0 + t * 16 + r + half * 8) * ldc + n;
      if (Cf) Cf[idx] = val;
      if (Cb) Cb[idx] = f2bf(val);
    }
  }
}

// ---------------------------------------------------------------------------
// Attention kernels
// ---------------------------------------------------------------------------
// One wave per (edge, head): score = dot(q[dst], k[src]+e) * scale;
// also atomicMax into per-(node,head) running max (ordered-uint encoding).
__global__ __launch_bounds__(256)
void attn_score(const float* __restrict__ q, const float* __restrict__ k,
                const float* __restrict__ eproj,
                const int* __restrict__ src, const int* __restrict__ dst,
                float* __restrict__ sbuf, unsigned* __restrict__ smax,
                int ch, float scale) {
  int gw   = (blockIdx.x * 256 + threadIdx.x) >> 5;
  int lane = threadIdx.x & 31;
  if (gw >= EE * HH) return;
  int e = gw >> 2, h = gw & 3;
  int s = src[e], d = dst[e];
  int D = HH * ch;
  const float* qp = q     + (size_t)d * D + h * ch;
  const float* kp = k     + (size_t)s * D + h * ch;
  const float* ep = eproj + (size_t)e * D + h * ch;
  float acc = 0.0f;
  for (int c = lane; c < ch; c += 32) acc += qp[c] * (kp[c] + ep[c]);
  for (int off = 16; off; off >>= 1) acc += __shfl_down(acc, off, 32);
  if (lane == 0) {
    float sc = acc * scale;
    sbuf[gw] = sc;
    atomicMax(&smax[d * HH + h], f2ord(sc));
  }
}

// a = exp(score - max[dst]); accumulate denominator per (node,head)
__global__ __launch_bounds__(256)
void attn_expsum(const float* __restrict__ sbuf, const unsigned* __restrict__ smax,
                 const int* __restrict__ dst, float* __restrict__ aexp,
                 float* __restrict__ ssum) {
  int i = blockIdx.x * 256 + threadIdx.x;
  if (i >= EE * HH) return;
  int e = i >> 2, h = i & 3;
  int d = dst[e];
  float a = __expf(sbuf[i] - ord2f(smax[d * HH + h]));
  aexp[i] = a;
  atomicAdd(&ssum[d * HH + h], a);
}

// One wave per (edge, head): agg[dst] += alpha * (v[src] + e)
__global__ __launch_bounds__(256)
void attn_message(const float* __restrict__ v, const float* __restrict__ eproj,
                  const float* __restrict__ aexp, const float* __restrict__ ssum,
                  const int* __restrict__ src, const int* __restrict__ dst,
                  float* __restrict__ agg, int ch) {
  int gw   = (blockIdx.x * 256 + threadIdx.x) >> 5;
  int lane = threadIdx.x & 31;
  if (gw >= EE * HH) return;
  int e = gw >> 2, h = gw & 3;
  int s = src[e], d = dst[e];
  int D = HH * ch;
  float alpha = aexp[gw] / ssum[d * HH + h];
  const float* vp = v     + (size_t)s * D + h * ch;
  const float* ep = eproj + (size_t)e * D + h * ch;
  float* ap = agg + (size_t)d * D + h * ch;
  for (int c = lane; c < ch; c += 32)
    atomicAdd(&ap[c], (vp[c] + ep[c]) * alpha);
}

// h = relu(bn(agg + skip)); write bf16 and/or f32
__global__ __launch_bounds__(256)
void node_update(const float* __restrict__ agg, const float* __restrict__ skip,
                 const float* __restrict__ gamma, const float* __restrict__ beta,
                 const float* __restrict__ mean,  const float* __restrict__ var,
                 unsigned short* __restrict__ out_bf, float* __restrict__ out_f,
                 int total, int D) {
  int idx = blockIdx.x * 256 + threadIdx.x;
  if (idx >= total) return;
  int d = idx % D;
  float x = agg[idx] + skip[idx];
  float y = gamma[d] * (x - mean[d]) * rsqrtf(var[d] + BN_EPS) + beta[d];
  y = fmaxf(y, 0.0f);
  if (out_bf) out_bf[idx] = f2bf(y);
  if (out_f)  out_f[idx]  = y;
}

// ---------------------------------------------------------------------------
// Readout kernels
// ---------------------------------------------------------------------------
__global__ __launch_bounds__(256)
void pool_add(const float* __restrict__ h2, const int* __restrict__ bids,
              float* __restrict__ g, float* __restrict__ cnt, int D) {
  int idx = blockIdx.x * 256 + threadIdx.x;
  if (idx >= NN * D) return;
  int n = idx / D, d = idx - n * D;
  int b = bids[n];
  atomicAdd(&g[(size_t)b * D + d], h2[idx]);
  if (d == 0) atomicAdd(&cnt[b], 1.0f);
}

// zcat[b, 0:1024] = bf16(g[b]/max(cnt,1))
__global__ __launch_bounds__(256)
void pool_finalize(const float* __restrict__ g, const float* __restrict__ cnt,
                   unsigned short* __restrict__ zcat, int D, int ldz) {
  int idx = blockIdx.x * 256 + threadIdx.x;
  if (idx >= BB * D) return;
  int b = idx / D, d = idx - b * D;
  float c = fmaxf(cnt[b], 1.0f);
  zcat[(size_t)b * ldz + d] = f2bf(g[idx] / c);
}

__global__ __launch_bounds__(256)
void bnrelu_fc(const float* __restrict__ in, const float* __restrict__ gamma,
               const float* __restrict__ beta, const float* __restrict__ mean,
               const float* __restrict__ var, float* __restrict__ out, int D) {
  int idx = blockIdx.x * 256 + threadIdx.x;
  if (idx >= BB * D) return;
  int d = idx % D;
  float y = gamma[d] * (in[idx] - mean[d]) * rsqrtf(var[d] + BN_EPS) + beta[d];
  out[idx] = fmaxf(y, 0.0f);
}

// out[b] = dot(z[b,0:128], W[:,0]) + bias; one wave per row
__global__ __launch_bounds__(256)
void fc2_dot(const float* __restrict__ z, const float* __restrict__ W,
             const float* __restrict__ b, float* __restrict__ out) {
  int gw   = (blockIdx.x * 256 + threadIdx.x) >> 5;
  int lane = threadIdx.x & 31;
  if (gw >= BB) return;
  const float* zp = z + (size_t)gw * 128;
  float acc = 0.0f;
  for (int c = lane; c < 128; c += 32) acc += zp[c] * W[c];
  for (int off = 16; off; off >>= 1) acc += __shfl_down(acc, off, 32);
  if (lane == 0) out[gw] = acc + b[0];
}

// ---------------------------------------------------------------------------
// Host launcher
// ---------------------------------------------------------------------------
static inline int cdiv(long long a, long long b) { return (int)((a + b - 1) / b); }

extern "C" void kernel_launch(void* const* d_in, const int* in_sizes, int n_in,
                              void* d_out, int out_size, void* d_ws, size_t ws_size,
                              hipStream_t stream) {
  (void)in_sizes; (void)n_in; (void)out_size; (void)ws_size;

  // ---- inputs (top-level insertion order; params flattened alphabetically) --
  const float* x        = (const float*)d_in[0];
  const int*   eidx     = (const int*)  d_in[1];   // [2,E]: row0=src, row1=dst
  const float* eattr    = (const float*)d_in[2];
  const int*   bids     = (const int*)  d_in[3];
  const float* solv     = (const float*)d_in[4];
  const float* bn1_beta = (const float*)d_in[5];
  const float* bn1_gam  = (const float*)d_in[6];
  const float* bn1_mean = (const float*)d_in[7];
  const float* bn1_var  = (const float*)d_in[8];
  const float* bn2_beta = (const float*)d_in[9];
  const float* bn2_gam  = (const float*)d_in[10];
  const float* bn2_mean = (const float*)d_in[11];
  const float* bn2_var  = (const float*)d_in[12];
  const float* bnf_beta = (const float*)d_in[13];
  const float* bnf_gam  = (const float*)d_in[14];
  const float* bnf_mean = (const float*)d_in[15];
  const float* bnf_var  = (const float*)d_in[16];
  const float* c1_We    = (const float*)d_in[17];
  const float* c1_Wk    = (const float*)d_in[18];
  const float* c1_Wq    = (const float*)d_in[19];
  const float* c1_Ws    = (const float*)d_in[20];
  const float* c1_Wv    = (const float*)d_in[21];
  const float* c1_bk    = (const float*)d_in[22];
  const float* c1_bq    = (const float*)d_in[23];
  const float* c1_bs    = (const float*)d_in[24];
  const float* c1_bv    = (const float*)d_in[25];
  const float* c2_We    = (const float*)d_in[26];
  const float* c2_Wk    = (const float*)d_in[27];
  const float* c2_Wq    = (const float*)d_in[28];
  const float* c2_Ws    = (const float*)d_in[29];
  const float* c2_Wv    = (const float*)d_in[30];
  const float* c2_bk    = (const float*)d_in[31];
  const float* c2_bq    = (const float*)d_in[32];
  const float* c2_bs    = (const float*)d_in[33];
  const float* c2_bv    = (const float*)d_in[34];
  const float* f1_W     = (const float*)d_in[35];
  const float* f1_b     = (const float*)d_in[36];
  const float* f2_W     = (const float*)d_in[37];
  const float* f2_b     = (const float*)d_in[38];
  const float* fs_W     = (const float*)d_in[39];
  const float* fs_b     = (const float*)d_in[40];

  const int* src = eidx;         // edge_index[0]
  const int* dst = eidx + EE;    // edge_index[1]
  float* out = (float*)d_out;

  // ---- workspace carve-up ----
  size_t off = 0;
  char* base = (char*)d_ws;
  auto take = [&](size_t bytes) -> char* {
    char* p = base + off;
    off += (bytes + 255) & ~(size_t)255;
    return p;
  };
  // packed bf16 weights
  unsigned short* Wt1e = (unsigned short*)take((size_t)512 * 32 * 2);
  unsigned short* Wt1k = (unsigned short*)take((size_t)512 * 64 * 2);
  unsigned short* Wt1q = (unsigned short*)take((size_t)512 * 64 * 2);
  unsigned short* Wt1s = (unsigned short*)take((size_t)512 * 64 * 2);
  unsigned short* Wt1v = (unsigned short*)take((size_t)512 * 64 * 2);
  unsigned short* Wt2e = (unsigned short*)take((size_t)1024 * 32 * 2);
  unsigned short* Wt2k = (unsigned short*)take((size_t)1024 * 512 * 2);
  unsigned short* Wt2q = (unsigned short*)take((size_t)1024 * 512 * 2);
  unsigned short* Wt2s = (unsigned short*)take((size_t)1024 * 512 * 2);
  unsigned short* Wt2v = (unsigned short*)take((size_t)1024 * 512 * 2);
  unsigned short* Wtso = (unsigned short*)take((size_t)128 * 128 * 2);
  unsigned short* Wtf1 = (unsigned short*)take((size_t)128 * 1152 * 2);
  // bf16 activations
  unsigned short* xb    = (unsigned short*)take((size_t)NN * 64 * 2);
  unsigned short* eb    = (unsigned short*)take((size_t)EE * 32 * 2);
  unsigned short* solvb = (unsigned short*)take((size_t)BB * 128 * 2);
  unsigned short* h1b   = (unsigned short*)take((size_t)NN * 512 * 2);
  unsigned short* zcat  = (unsigned short*)take((size_t)BB * 1152 * 2);
  // f32 big buffers (sized for conv2 widths; conv1 uses ldc=512 within them)
  float* qb    = (float*)take((size_t)NN * 1024 * 4);
  float* kb    = (float*)take((size_t)NN * 1024 * 4);
  float* vb    = (float*)take((size_t)NN * 1024 * 4);
  float* skipb = (float*)take((size_t)NN * 1024 * 4);
  float* eproj = (float*)take((size_t)EE * 1024 * 4);
  float* agg   = (float*)take((size_t)NN * 1024 * 4);
  float* h2    = eproj;          // alias: eproj dead after conv2 message pass
  // attention / readout small buffers
  float*    sbuf  = (float*)take((size_t)EE * HH * 4);
  float*    aexp  = (float*)take((size_t)EE * HH * 4);
  unsigned* smax  = (unsigned*)take((size_t)NN * HH * 4);
  float*    ssum  = (float*)take((size_t)NN * HH * 4);
  float*    gsum  = (float*)take((size_t)BB * 1024 * 4);
  float*    cnt   = (float*)take((size_t)BB * 4);
  float*    fc1o  = (float*)take((size_t)BB * 128 * 4);
  float*    zrelu = (float*)take((size_t)BB * 128 * 4);

  const dim3 T(256);
  auto ew = [](long long n) { return dim3((unsigned)cdiv(n, 256)); };

  // ---- pack weights + cast activations (cheap, deterministic each call) ----
  pack_wt_bf16<<<ew(512 * 32), T, 0, stream>>>(c1_We, Wt1e, 16, 512, 32);
  pack_wt_bf16<<<ew(512 * 64), T, 0, stream>>>(c1_Wk, Wt1k, 64, 512, 64);
  pack_wt_bf16<<<ew(512 * 64), T, 0, stream>>>(c1_Wq, Wt1q, 64, 512, 64);
  pack_wt_bf16<<<ew(512 * 64), T, 0, stream>>>(c1_Ws, Wt1s, 64, 512, 64);
  pack_wt_bf16<<<ew(512 * 64), T, 0, stream>>>(c1_Wv, Wt1v, 64, 512, 64);
  pack_wt_bf16<<<ew(1024 * 32), T, 0, stream>>>(c2_We, Wt2e, 16, 1024, 32);
  pack_wt_bf16<<<ew(1024 * 512), T, 0, stream>>>(c2_Wk, Wt2k, 512, 1024, 512);
  pack_wt_bf16<<<ew(1024 * 512), T, 0, stream>>>(c2_Wq, Wt2q, 512, 1024, 512);
  pack_wt_bf16<<<ew(1024 * 512), T, 0, stream>>>(c2_Ws, Wt2s, 512, 1024, 512);
  pack_wt_bf16<<<ew(1024 * 512), T, 0, stream>>>(c2_Wv, Wt2v, 512, 1024, 512);
  pack_wt_bf16<<<ew(128 * 128), T, 0, stream>>>(fs_W, Wtso, 128, 128, 128);
  pack_wt_bf16<<<ew(128 * 1152), T, 0, stream>>>(f1_W, Wtf1, 1152, 128, 1152);
  cast_bf16_pad<<<ew((long long)NN * 64), T, 0, stream>>>(x, xb, NN, 64, 64);
  cast_bf16_pad<<<ew((long long)EE * 32), T, 0, stream>>>(eattr, eb, EE, 16, 32);
  cast_bf16_pad<<<ew((long long)BB * 128), T, 0, stream>>>(solv, solvb, BB, 128, 128);

  // =========================== conv1 (ch=128, D=512) =======================
  {
    const int D = 512, ch = 128;
    dim3 gN(cdiv(D / 16, 8), NN / (16 * MT));
    wmma_gemm_bf16<<<gN, T, 0, stream>>>(xb, Wt1q, c1_bq, qb, nullptr, NN, D, 64, D, 0);
    wmma_gemm_bf16<<<gN, T, 0, stream>>>(xb, Wt1k, c1_bk, kb, nullptr, NN, D, 64, D, 0);
    wmma_gemm_bf16<<<gN, T, 0, stream>>>(xb, Wt1v, c1_bv, vb, nullptr, NN, D, 64, D, 0);
    wmma_gemm_bf16<<<gN, T, 0, stream>>>(xb, Wt1s, c1_bs, skipb, nullptr, NN, D, 64, D, 0);
    dim3 gE(cdiv(D / 16, 8), EE / (16 * MT));
    wmma_gemm_bf16<<<gE, T, 0, stream>>>(eb, Wt1e, nullptr, eproj, nullptr, EE, D, 32, D, 0);

    hipMemsetAsync(smax, 0, (size_t)NN * HH * 4, stream);
    hipMemsetAsync(ssum, 0, (size_t)NN * HH * 4, stream);
    hipMemsetAsync(agg,  0, (size_t)NN * D * 4, stream);

    attn_score<<<dim3(EE * HH / 8), T, 0, stream>>>(qb, kb, eproj, src, dst, sbuf, smax,
                                                    ch, rsqrtf((float)ch));
    attn_expsum<<<ew((long long)EE * HH), T, 0, stream>>>(sbuf, smax, dst, aexp, ssum);
    attn_message<<<dim3(EE * HH / 8), T, 0, stream>>>(vb, eproj, aexp, ssum, src, dst, agg, ch);
    node_update<<<ew((long long)NN * D), T, 0, stream>>>(
        agg, skipb, bn1_gam, bn1_beta, bn1_mean, bn1_var, h1b, nullptr, NN * D, D);
  }

  // =========================== conv2 (ch=256, D=1024) ======================
  {
    const int D = 1024, ch = 256;
    dim3 gN(cdiv(D / 16, 8), NN / (16 * MT));
    wmma_gemm_bf16<<<gN, T, 0, stream>>>(h1b, Wt2q, c2_bq, qb, nullptr, NN, D, 512, D, 0);
    wmma_gemm_bf16<<<gN, T, 0, stream>>>(h1b, Wt2k, c2_bk, kb, nullptr, NN, D, 512, D, 0);
    wmma_gemm_bf16<<<gN, T, 0, stream>>>(h1b, Wt2v, c2_bv, vb, nullptr, NN, D, 512, D, 0);
    wmma_gemm_bf16<<<gN, T, 0, stream>>>(h1b, Wt2s, c2_bs, skipb, nullptr, NN, D, 512, D, 0);
    dim3 gE(cdiv(D / 16, 8), EE / (16 * MT));
    wmma_gemm_bf16<<<gE, T, 0, stream>>>(eb, Wt2e, nullptr, eproj, nullptr, EE, D, 32, D, 0);

    hipMemsetAsync(smax, 0, (size_t)NN * HH * 4, stream);
    hipMemsetAsync(ssum, 0, (size_t)NN * HH * 4, stream);
    hipMemsetAsync(agg,  0, (size_t)NN * D * 4, stream);

    attn_score<<<dim3(EE * HH / 8), T, 0, stream>>>(qb, kb, eproj, src, dst, sbuf, smax,
                                                    ch, rsqrtf((float)ch));
    attn_expsum<<<ew((long long)EE * HH), T, 0, stream>>>(sbuf, smax, dst, aexp, ssum);
    attn_message<<<dim3(EE * HH / 8), T, 0, stream>>>(vb, eproj, aexp, ssum, src, dst, agg, ch);
    // h2 aliases eproj (eproj no longer read)
    node_update<<<ew((long long)NN * D), T, 0, stream>>>(
        agg, skipb, bn2_gam, bn2_beta, bn2_mean, bn2_var, nullptr, h2, NN * D, D);
  }

  // =========================== readout =====================================
  hipMemsetAsync(gsum, 0, (size_t)BB * 1024 * 4, stream);
  hipMemsetAsync(cnt,  0, (size_t)BB * 4, stream);
  pool_add<<<ew((long long)NN * 1024), T, 0, stream>>>(h2, bids, gsum, cnt, 1024);
  pool_finalize<<<ew((long long)BB * 1024), T, 0, stream>>>(gsum, cnt, zcat, 1024, 1152);

  // s = relu(solv @ Ws + bs) -> zcat[:, 1024:1152] (bf16, strided)
  wmma_gemm_bf16<<<dim3(1, BB / (16 * MT)), T, 0, stream>>>(
      solvb, Wtso, fs_b, nullptr, zcat + 1024, BB, 128, 128, 1152, 1);
  // fc1: [B,1152] x [1152,128]
  wmma_gemm_bf16<<<dim3(1, BB / (16 * MT)), T, 0, stream>>>(
      zcat, Wtf1, f1_b, fc1o, nullptr, BB, 128, 1152, 128, 0);
  bnrelu_fc<<<ew((long long)BB * 128), T, 0, stream>>>(
      fc1o, bnf_gam, bnf_beta, bnf_mean, bnf_var, zrelu, 128);
  fc2_dot<<<dim3(BB / 8), T, 0, stream>>>(zrelu, f2_W, f2_b, out);
}
